// PositionalKernelAttention_47648367182409
// MI455X (gfx1250) — matrix operand order repaired and checked by reading, not compiled, not run on hardware
//
#include <hip/hip_runtime.h>
#include <hip/hip_bf16.h>
#include <math.h>

// Problem constants (match the reference)
#define BB   8
#define TT   512
#define CC   512
#define HH   8
#define HKVV 4
#define DD   64
#define SMAX 128.0f
#define BT   (BB * TT)          // 4096
#define NDELTA (2 * TT - 1)     // 1023

typedef __attribute__((ext_vector_type(16))) _Float16 v16h;
typedef __attribute__((ext_vector_type(8)))  _Float16 v8h;
typedef __attribute__((ext_vector_type(4)))  _Float16 v4h;
typedef __attribute__((ext_vector_type(8)))  float    v8f;

// Swizzle helpers (CDNA5 ISA 7.12.2, wave32, 16-bit operands):
//  A 16x32: lane (half, m) element e holds K = e + 8*half + (e>=8 ? 8 : 0)
//    => K index c maps to half=(c>>3)&1, e=(c&7)+((c>>4)<<3)
//  B 32x16 as staged in this file (contiguous, not ISA order): element e holds K = e + 16*half
//    => K index k maps to half=k>>4, e=k&15
__device__ __forceinline__ int a_half(int c) { return (c >> 3) & 1; }
__device__ __forceinline__ int a_elem(int c) { return (c & 7) + ((c >> 4) << 3); }

// ---------------------------------------------------------------------------
// Workgroup-tiled GEMM: C[M,N] = A[M,K] @ B[K,N], fp32 in/out, f16 WMMA inner.
// 128 threads (4 waves); block computes 64x64 of C. Tiles staged in LDS in
// WMMA-fragment order, so operand loads are contiguous v16h (ds_load_b128 x2).
// Staging uses float4 global loads; A stores are packed 8-byte f16x4.
// ---------------------------------------------------------------------------
#define GM 64
#define GN 64
#define GKS 32

__global__ __launch_bounds__(128)
void tile_gemm_f16(const float* __restrict__ A, const float* __restrict__ B,
                   float* __restrict__ C, int M, int N, int K) {
    __shared__ __align__(32) _Float16 sa[GM][2][16];   // A, fragment order
    __shared__ __align__(32) _Float16 sb[4][32][16];   // B, fragment order

    const int tid  = threadIdx.x;
    const int wave = tid >> 5;               // 0..3 -> 16-row strip
    const int lane = tid & 31;
    const int half = lane >> 4;
    const int mn   = lane & 15;
    const int row0 = blockIdx.x * GM;
    const int col0 = blockIdx.y * GN;

    v8f acc[4] = {v8f{}, v8f{}, v8f{}, v8f{}};

    for (int k0 = 0; k0 < K; k0 += GKS) {
        // stage A 64x32: float4 loads (coalesced along K), packed f16x4 stores
#pragma unroll
        for (int t = 0; t < 4; ++t) {
            const int id = tid + t * 128;          // 512 tasks of 4 elements
            const int r  = id >> 3;
            const int c0 = (id & 7) * 4;
            const float4 f =
                *(const float4*)(&A[(size_t)(row0 + r) * K + k0 + c0]);
            v4h hv;
            hv[0] = (_Float16)f.x; hv[1] = (_Float16)f.y;
            hv[2] = (_Float16)f.z; hv[3] = (_Float16)f.w;
            *(v4h*)(&sa[r][c0 >> 4][c0 & 15]) = hv;
        }
        // stage B 32x64: float4 loads (coalesced along N), 4 b16 scatter stores
#pragma unroll
        for (int t = 0; t < 4; ++t) {
            const int id = tid + t * 128;          // 512 tasks of 4 elements
            const int k  = id >> 4;
            const int c0 = (id & 15) * 4;
            const float4 f =
                *(const float4*)(&B[(size_t)(k0 + k) * N + col0 + c0]);
            const int nt = c0 >> 4;
            const int rr = ((k >> 4) << 4);
            const int e  = k & 15;
            const int n0 = c0 & 15;
            sb[nt][rr + n0 + 0][e] = (_Float16)f.x;
            sb[nt][rr + n0 + 1][e] = (_Float16)f.y;
            sb[nt][rr + n0 + 2][e] = (_Float16)f.z;
            sb[nt][rr + n0 + 3][e] = (_Float16)f.w;
        }
        __syncthreads();

        // contiguous fragment loads (v16h = 2x ds_load_b128)
        const v16h a = *(const v16h*)(&sa[wave * 16 + mn][half][0]);
#pragma unroll
        for (int nt = 0; nt < 4; ++nt) {
            const v16h b = *(const v16h*)(&sb[nt][half * 16 + mn][0]);
            acc[nt] = __builtin_amdgcn_wmma_f32_16x16x32_f16(
                false, a, false, b, (short)0, acc[nt], false, false);
        }
        __syncthreads();
    }

    // C/D layout: VGPR r -> row r + 8*half, col = lane&15
#pragma unroll
    for (int nt = 0; nt < 4; ++nt) {
#pragma unroll
        for (int r = 0; r < 8; ++r) {
            C[(size_t)(row0 + wave * 16 + r + 8 * half) * N +
              col0 + nt * 16 + mn] = acc[nt][r];
        }
    }
}

// ---------------------------------------------------------------------------
// Normalize (optional) one head-row of D=64 and transpose into [B,heads,T,D].
// ---------------------------------------------------------------------------
__global__ __launch_bounds__(64)
void norm_pack(const float* __restrict__ in, float* __restrict__ out,
               int heads, int donorm) {
    const int rid = blockIdx.x;              // (b*T + t)*heads + hd
    const int hd  = rid % heads;
    const int bt  = rid / heads;
    const int t   = bt % TT;
    const int b   = bt / TT;
    const int d   = threadIdx.x;

    float val = in[(size_t)bt * heads * DD + hd * DD + d];

    __shared__ float red[DD];
    float scale = 1.0f;
    if (donorm) {
        red[d] = val * val;
        __syncthreads();
        for (int s = 32; s > 0; s >>= 1) {
            if (d < s) red[d] += red[d + s];
            __syncthreads();
        }
        scale = 8.0f * rsqrtf(red[0] + 1e-6f);   // sqrt(64) = 8
    }
    out[(((size_t)b * heads + hd) * TT + t) * DD + d] = val * scale;
}

// ---------------------------------------------------------------------------
// Precompute warped positional table Pd[h, delta + (T-1), d].
// ---------------------------------------------------------------------------
__global__ __launch_bounds__(64)
void build_pd(const float* __restrict__ P, const float* __restrict__ sigma,
              float* __restrict__ Pd) {
    const int rid = blockIdx.x;              // h * NDELTA + di
    const int h   = rid / NDELTA;
    const int di  = rid % NDELTA;
    const int d   = threadIdx.x;

    const float delta = (float)(di - (TT - 1));
    const float s  = fabsf(sigma[h]) + 1e-6f;
    const float dw = SMAX * tanhf(delta / s);
    const float dt = dw + SMAX;
    const float fl = floorf(dt);
    int lo = (int)fl;
    int hi = lo + 1;
    const float frac = dt - fl;
    lo = lo < 0 ? 0 : (lo > 256 ? 256 : lo);
    hi = hi < 0 ? 0 : (hi > 256 ? 256 : hi);

    const float* Ph = P + (size_t)h * 257 * DD;
    Pd[((size_t)h * NDELTA + di) * DD + d] =
        (1.0f - frac) * Ph[lo * DD + d] + frac * Ph[hi * DD + d];
}

// ---------------------------------------------------------------------------
// Fused flash-style positional-kernel attention.
// One wave per (b, h, 16-row query tile). Online softmax in fp32.
// Scores: 3-way contraction with Q, K and a 47-row LDS-staged Pd slab
// (delta = i-j+31 in [0,46]). V and softmax probs live in LDS in
// WMMA-fragment f16 order; PV via 4x v_wmma_f32_16x16x32_f16 per chunk.
// ---------------------------------------------------------------------------
__global__ __launch_bounds__(32)
void attn_kernel(const float* __restrict__ qn,   // [B,H,T,D]   (normalized)
                 const float* __restrict__ kn,   // [B,HKV,T,D] (normalized)
                 const float* __restrict__ vn,   // [B,HKV,T,D]
                 const float* __restrict__ Pd,   // [H,NDELTA,D]
                 float* __restrict__ yout) {     // [B,T,H*D]
    const int b  = blockIdx.x;
    const int h  = blockIdx.y;
    const int it = blockIdx.z;
    const int i0 = it * 16;

    const int lane = threadIdx.x;
    const int half = lane >> 4;
    const int ln   = lane & 15;

    __shared__ __align__(16) float qs[16][DD];
    __shared__ __align__(16) float ks[32][DD];
    __shared__ __align__(16) float pds[47][DD];        // Pd rows, local = i-j+31
    __shared__ __align__(32) _Float16 vsz[4][32][16];  // V, B-fragment order
    __shared__ __align__(32) _Float16 psz[32][16];     // probs, A-fragment order
    __shared__ float rowm[16], rowl[16], alpha[16];
    __shared__ float pmax[2][16], psums[2][16];

    const float* qbase = qn + (((size_t)b * HH + h) * TT + i0) * DD;
    const float* kbase = kn + (((size_t)b * HKVV + (h >> 1)) * TT) * DD;
    const float* vbase = vn + (((size_t)b * HKVV + (h >> 1)) * TT) * DD;
    const float* pdh   = Pd + (size_t)h * NDELTA * DD;

    {   // stage Q tile (float4 coalesced)
        const float4* qsrc = (const float4*)qbase;
        float4* qdst = (float4*)(&qs[0][0]);
        for (int idx = lane; idx < 16 * DD / 4; idx += 32) qdst[idx] = qsrc[idx];
    }
    if (lane < 16) { rowm[lane] = -3.0e38f; rowl[lane] = 0.0f; }
    __syncthreads();

    v8f acc[4] = {v8f{}, v8f{}, v8f{}, v8f{}};

    const int jmax = i0 + 15;
    for (int j0 = 0; j0 <= jmax; j0 += 32) {
        {   // stage K tile fp32 (float4), V tile f16 fragment-order,
            // and the 47-row Pd slab (float4)
            const float4* ksrc = (const float4*)(kbase + (size_t)j0 * DD);
            const float4* vsrc = (const float4*)(vbase + (size_t)j0 * DD);
            float4* kdst = (float4*)(&ks[0][0]);
            for (int idx = lane; idx < 32 * DD / 4; idx += 32) {
                kdst[idx] = ksrc[idx];
                const float4 vv = vsrc[idx];
                const int j  = (idx * 4) >> 6;       // V row (K dim of PV)
                const int d0 = (idx * 4) & 63;       // V col (N dim of PV)
                const int nt = d0 >> 4;
                const int rr = ((j >> 4) << 4);
                const int e  = j & 15;
                const int n0 = d0 & 15;
                vsz[nt][rr + n0 + 0][e] = (_Float16)vv.x;
                vsz[nt][rr + n0 + 1][e] = (_Float16)vv.y;
                vsz[nt][rr + n0 + 2][e] = (_Float16)vv.z;
                vsz[nt][rr + n0 + 3][e] = (_Float16)vv.w;
            }
            const int base = i0 - j0 - 31 + (TT - 1);
            const float4* psrc = (const float4*)(pdh + (size_t)base * DD);
            float4* pdst = (float4*)(&pds[0][0]);
            for (int idx = lane; idx < 47 * DD / 4; idx += 32) pdst[idx] = psrc[idx];
        }
        // prefetch next chunk's K/V (global_prefetch_b8 path)
        if (j0 + 32 <= jmax) {
            __builtin_prefetch(kbase + (size_t)(j0 + 32) * DD + lane * 64, 0, 1);
            __builtin_prefetch(vbase + (size_t)(j0 + 32) * DD + lane * 64, 0, 1);
        }
        __syncthreads();

        // scores: lane handles row i = ln, cols j = 16*half + jj
        const int i  = ln;
        const int gi = i0 + i;
        float svals[16];
        float smax = -3.0e38f;
#pragma unroll 1
        for (int jj = 0; jj < 16; ++jj) {
            const int j  = half * 16 + jj;
            const int gj = j0 + j;
            const float* pr = &pds[i - j + 31][0];   // delta row in LDS
            float s = 0.0f;
#pragma unroll 4
            for (int d = 0; d < DD; ++d)
                s = fmaf(qs[i][d] * pr[d], ks[j][d], s);
            s *= 0.125f;                     // 1/sqrt(D)
            if (gj > gi) s = -3.0e38f;       // causal mask
            svals[jj] = s;
            smax = fmaxf(smax, s);
        }
        pmax[half][i] = smax;
        __syncthreads();

        if (lane < 16) {
            const float mold = rowm[lane];
            const float mnew = fmaxf(mold, fmaxf(pmax[0][lane], pmax[1][lane]));
            alpha[lane] = __expf(mold - mnew);
            rowm[lane]  = mnew;
        }
        __syncthreads();

        // probs: columns jj=0..7 and 8..15 map to two CONTIGUOUS 8-f16 runs
        // of the A-fragment buffer -> two 16-byte vector stores per lane.
        //   lane (half, i): psz[16*half+i][0..7]  <- jj = 0..7
        //                   psz[16*half+i][8..15] <- jj = 8..15
        const float mnew = rowm[i];
        float psum = 0.0f;
        v8h p0, p1;
#pragma unroll
        for (int jj = 0; jj < 8; ++jj) {
            float p = __expf(svals[jj] - mnew);
            if (svals[jj] <= -1.0e38f) p = 0.0f;
            p0[jj] = (_Float16)p;
            psum += p;
        }
#pragma unroll
        for (int jj = 8; jj < 16; ++jj) {
            float p = __expf(svals[jj] - mnew);
            if (svals[jj] <= -1.0e38f) p = 0.0f;
            p1[jj - 8] = (_Float16)p;
            psum += p;
        }
        *(v8h*)(&psz[16 * half + i][0]) = p0;
        *(v8h*)(&psz[16 * half + i][8]) = p1;
        psums[half][i] = psum;
        __syncthreads();

        if (lane < 16)
            rowl[lane] = rowl[lane] * alpha[lane] + psums[0][lane] + psums[1][lane];

        // rescale accumulators by alpha (per output row m = r + 8*half)
        float av[8];
#pragma unroll
        for (int r = 0; r < 8; ++r) av[r] = alpha[r + 8 * half];
#pragma unroll
        for (int r = 0; r < 8; ++r) {
            acc[0][r] *= av[r]; acc[1][r] *= av[r];
            acc[2][r] *= av[r]; acc[3][r] *= av[r];
        }

        // PV: acc[16 x 64] += P[16 x 32] @ V[32 x 64]; contiguous frag loads
        const v16h afrag = *(const v16h*)(&psz[lane][0]);
#pragma unroll
        for (int t = 0; t < 4; ++t) {
            const v16h bfrag = *(const v16h*)(&vsz[t][lane][0]);
            acc[t] = __builtin_amdgcn_wmma_f32_16x16x32_f16(
                false, afrag, false, bfrag, (short)0, acc[t], false, false);
        }
        __syncthreads();   // protect ks/vsz/psz/pds before next iteration
    }

    // finalize: y = acc / l, store to [B,T,H*D]
    __syncthreads();
#pragma unroll
    for (int t = 0; t < 4; ++t) {
#pragma unroll
        for (int r = 0; r < 8; ++r) {
            const int m = r + 8 * half;
            const int d = t * 16 + ln;
            yout[((size_t)b * TT + i0 + m) * (HH * DD) + h * DD + d] =
                acc[t][r] / rowl[m];
        }
    }
}

// ---------------------------------------------------------------------------
// Host-side orchestration
// ---------------------------------------------------------------------------
extern "C" void kernel_launch(void* const* d_in, const int* in_sizes, int n_in,
                              void* d_out, int out_size, void* d_ws, size_t ws_size,
                              hipStream_t stream) {
    const float* x     = (const float*)d_in[0];   // [B,T,C]
    const float* Wq    = (const float*)d_in[1];   // [C, H*D]
    const float* Wk    = (const float*)d_in[2];   // [C, HKV*D]
    const float* Wv    = (const float*)d_in[3];   // [C, HKV*D]
    const float* Wproj = (const float*)d_in[4];   // [C, C]
    const float* P     = (const float*)d_in[5];   // [H, 257, D]
    const float* sigma = (const float*)d_in[6];   // [H]
    float*       out   = (float*)d_out;           // [B,T,C]

    // Workspace carve-out (fp32)
    float* ws = (float*)d_ws;
    float* qf  = ws;                              // [BT, H*D]
    float* kf  = qf  + (size_t)BT * HH * DD;      // [BT, HKV*D]
    float* vf  = kf  + (size_t)BT * HKVV * DD;    // [BT, HKV*D]
    float* qn  = vf  + (size_t)BT * HKVV * DD;    // [B,H,T,D]
    float* kn  = qn  + (size_t)BB * HH * TT * DD; // [B,HKV,T,D]
    float* vn  = kn  + (size_t)BB * HKVV * TT * DD;
    float* pd  = vn  + (size_t)BB * HKVV * TT * DD; // [H,NDELTA,D]
    float* yat = pd  + (size_t)HH * NDELTA * DD;  // [B,T,H*D]

    // 1) QKV projections (WMMA GEMMs, LDS-tiled)
    tile_gemm_f16<<<dim3(BT / GM, (HH * DD) / GN), 128, 0, stream>>>(
        x, Wq, qf, BT, HH * DD, CC);
    tile_gemm_f16<<<dim3(BT / GM, (HKVV * DD) / GN), 128, 0, stream>>>(
        x, Wk, kf, BT, HKVV * DD, CC);
    tile_gemm_f16<<<dim3(BT / GM, (HKVV * DD) / GN), 128, 0, stream>>>(
        x, Wv, vf, BT, HKVV * DD, CC);

    // 2) normalize q,k; transpose v into [B,heads,T,D]
    norm_pack<<<BT * HH,   64, 0, stream>>>(qf, qn, HH,   1);
    norm_pack<<<BT * HKVV, 64, 0, stream>>>(kf, kn, HKVV, 1);
    norm_pack<<<BT * HKVV, 64, 0, stream>>>(vf, vn, HKVV, 0);

    // 3) warped positional table Pd[h, delta, d]
    build_pd<<<HH * NDELTA, 64, 0, stream>>>(P, sigma, pd);

    // 4) fused attention (VALU scores + WMMA PV)
    attn_kernel<<<dim3(BB, HH, TT / 16), 32, 0, stream>>>(qn, kn, vn, pd, yat);

    // 5) output projection (WMMA GEMM) straight into d_out
    tile_gemm_f16<<<dim3(BT / GM, CC / GN), 128, 0, stream>>>(
        yat, Wproj, out, BT, CC, CC);
}
